// EuclideanCodebook_18047452578777
// MI455X (gfx1250) — matrix operand order repaired and checked by reading, not compiled, not run on hardware
//
#include <hip/hip_runtime.h>

#define DIM        128
#define KCB        1024
#define TILE_ROWS  32    // two 16-row A sets share each loaded B fragment
#define WAVES      8
#define BLOCK      256
#define LDS_STRIDE 132   // 132 % 64 == 4 -> conflict-free strided row reads
#define KSTEPS     4     // 128 / 32 K-steps per 16x16x32 WMMA chain

typedef _Float16 half_t;
typedef __attribute__((ext_vector_type(16))) _Float16 v16h;
typedef __attribute__((ext_vector_type(8)))  float    v8f;

// ---- kernel 0: e_sq + split embed into f16 hi/lo ---------------------------
__global__ __launch_bounds__(256)
void vq_prep_kernel(const float* __restrict__ embed,
                    float* __restrict__ esq,
                    half_t* __restrict__ ehi,
                    half_t* __restrict__ elo) {
    int k = blockIdx.x * blockDim.x + threadIdx.x;
    if (k >= KCB) return;
    const float4* row = (const float4*)(embed + (size_t)k * DIM);
    float s = 0.f;
#pragma unroll
    for (int i = 0; i < DIM / 4; ++i) {
        float4 v = row[i];
        s += v.x * v.x + v.y * v.y + v.z * v.z + v.w * v.w;
        float f[4] = {v.x, v.y, v.z, v.w};
#pragma unroll
        for (int j = 0; j < 4; ++j) {
            half_t h = (half_t)f[j];
            ehi[(size_t)k * DIM + 4 * i + j] = h;
            elo[(size_t)k * DIM + 4 * i + j] = (half_t)(f[j] - (float)h);
        }
    }
    esq[k] = s;
}

// build one row-set of A fragments (hi/lo) from the LDS-staged x tile
__device__ __forceinline__ void build_a_frags(const float* __restrict__ xrow, int hf,
                                              v16h (&Ah)[KSTEPS], v16h (&Al)[KSTEPS]) {
#pragma unroll
    for (int kk = 0; kk < KSTEPS; ++kk) {
        const float* p0 = xrow + 32 * kk + 16 * hf;
#pragma unroll
        for (int e = 0; e < 8; ++e) {
            float f0 = p0[e];
            float f1 = p0[8 + e];
            half_t h0 = (half_t)f0;
            half_t h1 = (half_t)f1;
            Ah[kk][e]     = h0;  Al[kk][e]     = (half_t)(f0 - (float)h0);
            Ah[kk][8 + e] = h1;  Al[kk][8 + e] = (half_t)(f1 - (float)h1);
        }
    }
}

// ---- kernel 1: fused split-f16 WMMA GEMM + argmax + gather -----------------
__global__ __launch_bounds__(BLOCK)
void vq_search_kernel(const float* __restrict__ x,
                      const float* __restrict__ embed,
                      const float* __restrict__ esq,
                      const half_t* __restrict__ ehi,
                      const half_t* __restrict__ elo,
                      float* __restrict__ out_q,
                      float* __restrict__ out_i) {
    __shared__ float xs[TILE_ROWS * LDS_STRIDE];
    __shared__ float wbest[WAVES][TILE_ROWS];
    __shared__ int   widx [WAVES][TILE_ROWS];
    __shared__ int   fidx [TILE_ROWS];

    const int tid  = threadIdx.x;
    const int wave = tid >> 5;
    const int lane = tid & 31;
    const int nl   = lane & 15;   // M (for A/C) or N (for B/C) within half-wave
    const int hf   = lane >> 4;   // K sub-block selector
    const long row_base = (long)blockIdx.x * TILE_ROWS;

    // stage the 32x128 x tile into LDS (padded rows)
    for (int i = tid; i < TILE_ROWS * DIM; i += BLOCK) {
        int r = i >> 7, d = i & (DIM - 1);
        xs[r * LDS_STRIDE + d] = x[(row_base + r) * DIM + d];
    }
    __syncthreads();

    // two 16-row A sets, each split into f16 hi/lo (128 VGPRs total)
    v16h Ah0[KSTEPS], Al0[KSTEPS], Ah1[KSTEPS], Al1[KSTEPS];
    build_a_frags(&xs[nl * LDS_STRIDE],        hf, Ah0, Al0);
    build_a_frags(&xs[(nl + 16) * LDS_STRIDE], hf, Ah1, Al1);

    float bv0[8], bv1[8];
    int   bi0[8], bi1[8];
#pragma unroll
    for (int j = 0; j < 8; ++j) {
        bv0[j] = -3.4e38f; bi0[j] = 0;
        bv1[j] = -3.4e38f; bi1[j] = 0;
    }

    // this wave's 8 column tiles: col0(i) = 16*wave + 128*i
#pragma unroll
    for (int i = 0; i < 8; ++i) {
        const int col = 16 * wave + 128 * i + nl;
        const float e2 = esq[col];

        // B fragments: lane holds column `col`, K = 32*kk + 16*hf + [0,16)
        const half_t* bh = ehi + (size_t)col * DIM + 16 * hf;
        const half_t* bl = elo + (size_t)col * DIM + 16 * hf;

        v8f c0 = {}, c1 = {};
        // K split in halves to keep only 2 K-steps of B (32 VGPRs) live
#pragma unroll
        for (int kh = 0; kh < 2; ++kh) {
            v16h BH[2], BL[2];
#pragma unroll
            for (int t = 0; t < 2; ++t) {
                BH[t] = *(const v16h*)(bh + 32 * (2 * kh + t));
                BL[t] = *(const v16h*)(bl + 32 * (2 * kh + t));
            }
            // s = Ah*Bh + Ah*Bl + Al*Bh, both row sets share each B fragment
#pragma unroll
            for (int t = 0; t < 2; ++t) {
                const int kk = 2 * kh + t;
                c0 = __builtin_amdgcn_wmma_f32_16x16x32_f16(false, Ah0[kk], false, BH[t],
                                                            (short)0, c0, false, false);
                c1 = __builtin_amdgcn_wmma_f32_16x16x32_f16(false, Ah1[kk], false, BH[t],
                                                            (short)0, c1, false, false);
                c0 = __builtin_amdgcn_wmma_f32_16x16x32_f16(false, Ah0[kk], false, BL[t],
                                                            (short)0, c0, false, false);
                c1 = __builtin_amdgcn_wmma_f32_16x16x32_f16(false, Ah1[kk], false, BL[t],
                                                            (short)0, c1, false, false);
                c0 = __builtin_amdgcn_wmma_f32_16x16x32_f16(false, Al0[kk], false, BH[t],
                                                            (short)0, c0, false, false);
                c1 = __builtin_amdgcn_wmma_f32_16x16x32_f16(false, Al1[kk], false, BH[t],
                                                            (short)0, c1, false, false);
            }
        }

#pragma unroll
        for (int j = 0; j < 8; ++j) {
            float s0 = 2.f * c0[j] - e2;         // row j+8*hf,      col nl
            float s1 = 2.f * c1[j] - e2;         // row 16+j+8*hf,   col nl
            if (s0 > bv0[j]) { bv0[j] = s0; bi0[j] = col; }
            if (s1 > bv1[j]) { bv1[j] = s1; bi1[j] = col; }
        }
    }

    // reduce the 16 candidate columns inside each half-wave (masks stay < 16)
#pragma unroll
    for (int j = 0; j < 8; ++j) {
#pragma unroll
        for (int m = 8; m >= 1; m >>= 1) {
            float ov0 = __shfl_xor(bv0[j], m, 32);
            int   oi0 = __shfl_xor(bi0[j], m, 32);
            if (ov0 > bv0[j] || (ov0 == bv0[j] && oi0 < bi0[j])) { bv0[j] = ov0; bi0[j] = oi0; }
            float ov1 = __shfl_xor(bv1[j], m, 32);
            int   oi1 = __shfl_xor(bi1[j], m, 32);
            if (ov1 > bv1[j] || (ov1 == bv1[j] && oi1 < bi1[j])) { bv1[j] = ov1; bi1[j] = oi1; }
        }
        if (nl == 0) {
            int r0 = j + 8 * hf;
            wbest[wave][r0]      = bv0[j];
            widx [wave][r0]      = bi0[j];
            wbest[wave][16 + r0] = bv1[j];
            widx [wave][16 + r0] = bi1[j];
        }
    }
    __syncthreads();

    // cross-wave reduction + index output
    if (tid < TILE_ROWS) {
        float best = wbest[0][tid];
        int   bidx = widx [0][tid];
#pragma unroll
        for (int w = 1; w < WAVES; ++w) {
            float v  = wbest[w][tid];
            int   ii = widx [w][tid];
            if (v > best || (v == best && ii < bidx)) { best = v; bidx = ii; }
        }
        fidx[tid] = bidx;
        out_i[row_base + tid] = (float)bidx;
    }
    __syncthreads();

    // gather quantized = embed[best_idx] (exact f32 rows) with float4 stores
    const float4* ef = (const float4*)embed;
    float4*       of = (float4*)out_q;
    for (int i = tid; i < TILE_ROWS * (DIM / 4); i += BLOCK) {
        int r = i >> 5, d4 = i & 31;
        of[(row_base + r) * (DIM / 4) + d4] = ef[(size_t)fidx[r] * (DIM / 4) + d4];
    }
}

extern "C" void kernel_launch(void* const* d_in, const int* in_sizes, int n_in,
                              void* d_out, int out_size, void* d_ws, size_t ws_size,
                              hipStream_t stream) {
    const float* x     = (const float*)d_in[0];
    const float* embed = (const float*)d_in[1];

    // ws layout: esq (4 KB) | ehi (256 KB f16) | elo (256 KB f16)
    float*  esq = (float*)d_ws;
    half_t* ehi = (half_t*)((char*)d_ws + KCB * sizeof(float));
    half_t* elo = ehi + (size_t)KCB * DIM;

    const int nrows = in_sizes[0] / DIM;                // 65536
    float* out_q = (float*)d_out;                       // (N, 128) quantized
    float* out_i = (float*)d_out + (size_t)nrows * DIM; // (N,) indices

    vq_prep_kernel<<<(KCB + 255) / 256, 256, 0, stream>>>(embed, esq, ehi, elo);
    vq_search_kernel<<<nrows / TILE_ROWS, BLOCK, 0, stream>>>(x, embed, esq,
                                                              ehi, elo,
                                                              out_q, out_i);
}
